// LSTM_21973052686713
// MI455X (gfx1250) — matrix operand order repaired and checked by reading, not compiled, not run on hardware
//
#include <hip/hip_runtime.h>
#include <hip/hip_bf16.h>
#include <math.h>

// ---------------------------------------------------------------------------
// Types for CDNA5 WMMA
// ---------------------------------------------------------------------------
typedef __attribute__((ext_vector_type(16))) __bf16 v16bf;
typedef __attribute__((ext_vector_type(8)))  float  v8f;

#define Hdim 1024
#define Nbat 256
#define Lseq 512
#define GATES 4096   // 4*Hdim

// ---------------------------------------------------------------------------
// Workspace layout (bytes)
// ---------------------------------------------------------------------------
#define OFF_WHH0P   0u
#define OFF_WIH1P   (OFF_WHH0P + 4096u*1024u*2u)     //  8 MB
#define OFF_WHH1P   (OFF_WIH1P + 4096u*1024u*2u)     //  8 MB
#define OFF_BIAS0   (OFF_WHH1P + 4096u*1024u*2u)     //  8 MB
#define OFF_BIAS1   (OFF_BIAS0 + 4096u*4u)
#define OFF_H0P0    (OFF_BIAS1 + 4096u*4u)           // zero region starts here
#define OFF_H0P1    (OFF_H0P0 + 256u*1024u*2u)
#define OFF_H1P0    (OFF_H0P1 + 256u*1024u*2u)
#define OFF_H1P1    (OFF_H1P0 + 256u*1024u*2u)
#define OFF_C0      (OFF_H1P1 + 256u*1024u*2u)
#define OFF_C1      (OFF_C0   + 256u*1024u*4u)
#define OFF_HF1     (OFF_C1   + 256u*1024u*4u)
#define ZERO_FLOATS ((OFF_HF1 + 256u*1024u*4u - OFF_H0P0) / 4u)

// ---------------------------------------------------------------------------
// CDNA5 async global -> LDS copy (16 B per lane per issue), tracked by
// ASYNCcnt. Generic LDS pointer's low 32 bits are the LDS byte offset.
// ---------------------------------------------------------------------------
__device__ __forceinline__ void g2l_async_b128(void* lds_ptr, const void* gptr) {
    unsigned lds_off = (unsigned)(size_t)lds_ptr;
    unsigned long long ga = (unsigned long long)(size_t)gptr;
    asm volatile("global_load_async_to_lds_b128 %0, %1, off"
                 :: "v"(lds_off), "v"(ga)
                 : "memory");
}
__device__ __forceinline__ void wait_async0() {
    asm volatile("s_wait_asynccnt 0x0" ::: "memory");
}

// ---------------------------------------------------------------------------
// Pack a (4096 x K) fp32 row-major weight into bf16 WMMA B-fragment tiles.
// B tile = 32 (K) x 16 (N); per-lane v16bf stored contiguously (32 B / lane).
// Lane l, element e -> (K = (e&7) + 8*(l>>4) + 16*(e>>3), N = l&15) within the tile.
// Tile index = colTile * (K/32) + kc ; flat elem = tile*512 + l*16 + e.
// ---------------------------------------------------------------------------
__global__ void pack_weight_bf16(const float* __restrict__ W,
                                 __bf16* __restrict__ P, int K) {
    int idx = blockIdx.x * blockDim.x + threadIdx.x;
    int total = GATES * K;
    if (idx >= total) return;
    int numKc  = K >> 5;
    int tile   = idx >> 9;        // /512
    int within = idx & 511;
    int l = within >> 4;
    int e = within & 15;
    int ctg = tile / numKc;
    int kc  = tile % numKc;
    int n   = l & 15;
    int kt  = (e & 7) + ((l >> 4) << 3) + ((e >> 3) << 4);
    int g   = ctg * 16 + n;
    int k   = kc * 32 + kt;
    P[idx] = (__bf16)W[(size_t)g * K + k];
}

__global__ void make_bias(const float* __restrict__ bih0, const float* __restrict__ bhh0,
                          const float* __restrict__ bih1, const float* __restrict__ bhh1,
                          float* __restrict__ b0, float* __restrict__ b1) {
    int i = blockIdx.x * blockDim.x + threadIdx.x;
    if (i < GATES) {
        b0[i] = bih0[i] + bhh0[i];
        b1[i] = bih1[i] + bhh1[i];
    }
}

__global__ void zero_f32(float* __restrict__ p, unsigned n) {
    unsigned i = blockIdx.x * blockDim.x + threadIdx.x;
    if (i < n) p[i] = 0.0f;
}

__device__ __forceinline__ float sigf(float x) {
    return 1.0f / (1.0f + __expf(-x));
}

// ---------------------------------------------------------------------------
// One GEMM pass over K=1024: A fragments from LDS (ds_load_b128), B fragments
// from the L2-resident packed weights. All loads issued before the 4 WMMAs;
// #pragma unroll 1 keeps one accumulator set (no dual_mov churn).
// ---------------------------------------------------------------------------
__device__ __forceinline__ void gemm_pass(const __bf16* smA,
                                          const __bf16* __restrict__ Wp,
                                          int ct, int lane,
                                          v8f& a0, v8f& a1, v8f& a2, v8f& a3) {
    const size_t GSTRIDE = (size_t)64 * 32 * 512;   // one gate block (2 MB)
    #pragma unroll 1
    for (int kc = 0; kc < 32; ++kc) {
        const __bf16* wb = Wp + ((size_t)ct * 32 + kc) * 512 + lane * 16;
        v16bf b0 = *(const v16bf*)(wb);
        v16bf b1 = *(const v16bf*)(wb + GSTRIDE);
        v16bf b2 = *(const v16bf*)(wb + 2 * GSTRIDE);
        v16bf b3 = *(const v16bf*)(wb + 3 * GSTRIDE);
        v16bf a  = *(const v16bf*)(smA + kc * 512 + lane * 16);
        if (kc + 1 < 32)
            __builtin_prefetch(wb + 512, 0, 1);
        a0 = __builtin_amdgcn_wmma_f32_16x16x32_bf16(false, a, false, b0, (short)0, a0, false, false);
        a1 = __builtin_amdgcn_wmma_f32_16x16x32_bf16(false, a, false, b1, (short)0, a1, false, false);
        a2 = __builtin_amdgcn_wmma_f32_16x16x32_bf16(false, a, false, b2, (short)0, a2, false, false);
        a3 = __builtin_amdgcn_wmma_f32_16x16x32_bf16(false, a, false, b3, (short)0, a3, false, false);
    }
}

// ---------------------------------------------------------------------------
// One LSTM timestep. 1024 waves: gw = rt*64+ct ; rt = batch tile (16 rows),
// ct = hidden-column tile (16 cols). Each wave accumulates the 4 gate tiles
// (i,f,g,o) at the same hidden column so the cell update is register-local.
// Block = 256 threads (8 waves); all 8 waves of a block share rt, so the
// packed A row-tile (32 KB) is staged once per block via async global->LDS.
// ---------------------------------------------------------------------------
template <int LAYER>
__global__ void __launch_bounds__(256)
lstm_step(const __bf16* __restrict__ hprev_p,   // packed A-frag h_{t-1} (256x1024)
          __bf16* __restrict__ hnext_p,         // packed A-frag h_t
          const __bf16* __restrict__ xp,        // packed A-frag x_t (layer1)
          const __bf16* __restrict__ Whh_p,     // packed B-frag recurrent weights
          const __bf16* __restrict__ Wx_p,      // packed B-frag input weights (layer1)
          const float* __restrict__ bias,       // combined bias (4096)
          const float* __restrict__ seq,        // (L,N,6) fp32 (layer0)
          const float* __restrict__ Wih0,       // (4096,6) fp32 (layer0)
          float* __restrict__ c,                // (256,1024) fp32 cell state
          float* __restrict__ hf,               // (256,1024) fp32 h out (layer1)
          int t) {
    const int lane = threadIdx.x & 31;
    const int wave = threadIdx.x >> 5;
    const int gw   = blockIdx.x * 8 + wave;
    const int rt   = gw >> 6;    // 0..15 batch tile (uniform per block)
    const int ct   = gw & 63;    // 0..63 hidden column tile
    const int lhi8 = (lane >> 4) << 3;   // 0 or 8

    __shared__ __align__(16) __bf16 smA[32 * 512];                  // h_{t-1} row tile
    __shared__ __align__(16) __bf16 smX[(LAYER == 1) ? 32 * 512 : 16];  // x_t row tile
    __shared__ float sx[16 * 6];

    // ---- async-stage the packed A row-tile(s) into LDS (32 KB each) ----
    {
        const char* gh = (const char*)(hprev_p + (size_t)rt * 16384);
        char* lh = (char*)smA + threadIdx.x * 128;
        #pragma unroll
        for (int j = 0; j < 8; ++j)
            g2l_async_b128(lh + j * 16, gh + threadIdx.x * 128 + j * 16);
        if (LAYER == 1) {
            const char* gx = (const char*)(xp + (size_t)rt * 16384);
            char* lx = (char*)smX + threadIdx.x * 128;
            #pragma unroll
            for (int j = 0; j < 8; ++j)
                g2l_async_b128(lx + j * 16, gx + threadIdx.x * 128 + j * 16);
        }
        if (LAYER == 0 && threadIdx.x < 96) {
            int nl = threadIdx.x / 6, k = threadIdx.x % 6;
            sx[threadIdx.x] = seq[((size_t)t * Nbat + rt * 16 + nl) * 6 + k];
        }
        wait_async0();
        __syncthreads();
    }

    v8f acc0 = (v8f){0.f,0.f,0.f,0.f,0.f,0.f,0.f,0.f};
    v8f acc1 = acc0, acc2 = acc0, acc3 = acc0;

    // ---- recurrent GEMM: gates += h_{t-1} @ Whh^T (K = 1024) ----
    gemm_pass(smA, Whh_p, ct, lane, acc0, acc1, acc2, acc3);

    // ---- layer1: fused input projection gates += x_t @ Wih1^T ----
    if (LAYER == 1)
        gemm_pass(smX, Wx_p, ct, lane, acc0, acc1, acc2, acc3);

    // ---- bias (+ layer0 K=6 input projection via VALU) ----
    float badd[4];
    #pragma unroll
    for (int g = 0; g < 4; ++g)
        badd[g] = bias[g * Hdim + ct * 16 + (lane & 15)];

    v8f accs[4] = { acc0, acc1, acc2, acc3 };
    if (LAYER == 0) {
        #pragma unroll
        for (int g = 0; g < 4; ++g) {
            const float* wr = &Wih0[(size_t)(g * Hdim + ct * 16 + (lane & 15)) * 6];
            float w0 = wr[0], w1 = wr[1], w2 = wr[2], w3 = wr[3], w4 = wr[4], w5 = wr[5];
            #pragma unroll
            for (int v = 0; v < 8; ++v) {
                const float* sr = &sx[(v + lhi8) * 6];
                accs[g][v] += w0*sr[0] + w1*sr[1] + w2*sr[2] + w3*sr[3] + w4*sr[4] + w5*sr[5];
            }
        }
    }

    // ---- nonlinearity + cell/hidden update (register-local per lane) ----
    const int jcol = ct * 16 + (lane & 15);
    #pragma unroll
    for (int v = 0; v < 8; ++v) {
        int n = rt * 16 + v + lhi8;
        float gi = sigf(accs[0][v] + badd[0]);
        float gf = sigf(accs[1][v] + badd[1]);
        float gg = tanhf(accs[2][v] + badd[2]);
        float go = sigf(accs[3][v] + badd[3]);
        size_t idx = (size_t)n * Hdim + jcol;
        float cn = gf * c[idx] + gi * gg;
        c[idx] = cn;
        float h = go * tanhf(cn);
        if (LAYER == 1) hf[idx] = h;
        // scatter h into next step's packed A-fragment layout
        int kc  = jcol >> 5;
        int kin = jcol & 31;
        int grp = kin >> 3;
        int e   = (kin & 7) + ((grp >> 1) << 3);
        int lp  = ((grp & 1) << 4) + (v + lhi8);
        hnext_p[((size_t)rt * 32 + kc) * 512 + lp * 16 + e] = (__bf16)h;
    }
}

// ---------------------------------------------------------------------------
// Final linear head: out[n] = h1[n,:] . lin_W + lin_b   (256 outputs)
// ---------------------------------------------------------------------------
__global__ void final_linear(const float* __restrict__ hf,
                             const float* __restrict__ linW,
                             const float* __restrict__ linb,
                             float* __restrict__ out) {
    int n = blockIdx.x * blockDim.x + threadIdx.x;
    if (n >= Nbat) return;
    float s = linb[0];
    const float* hr = hf + (size_t)n * Hdim;
    for (int k = 0; k < Hdim; ++k) s += hr[k] * linW[k];
    out[n] = s;
}

// ---------------------------------------------------------------------------
extern "C" void kernel_launch(void* const* d_in, const int* in_sizes, int n_in,
                              void* d_out, int out_size, void* d_ws, size_t ws_size,
                              hipStream_t stream) {
    const float* seq   = (const float*)d_in[0];
    const float* Wih0  = (const float*)d_in[1];
    const float* Whh0  = (const float*)d_in[2];
    const float* bih0  = (const float*)d_in[3];
    const float* bhh0  = (const float*)d_in[4];
    const float* Wih1  = (const float*)d_in[5];
    const float* Whh1  = (const float*)d_in[6];
    const float* bih1  = (const float*)d_in[7];
    const float* bhh1  = (const float*)d_in[8];
    const float* linW  = (const float*)d_in[9];
    const float* linb  = (const float*)d_in[10];
    float* out = (float*)d_out;

    char* ws = (char*)d_ws;
    __bf16* Whh0p = (__bf16*)(ws + OFF_WHH0P);
    __bf16* Wih1p = (__bf16*)(ws + OFF_WIH1P);
    __bf16* Whh1p = (__bf16*)(ws + OFF_WHH1P);
    float*  b0    = (float*)(ws + OFF_BIAS0);
    float*  b1    = (float*)(ws + OFF_BIAS1);
    __bf16* h0p[2] = { (__bf16*)(ws + OFF_H0P0), (__bf16*)(ws + OFF_H0P1) };
    __bf16* h1p[2] = { (__bf16*)(ws + OFF_H1P0), (__bf16*)(ws + OFF_H1P1) };
    float*  c0    = (float*)(ws + OFF_C0);
    float*  c1    = (float*)(ws + OFF_C1);
    float*  hf1   = (float*)(ws + OFF_HF1);

    // one-time prep (re-run every call: deterministic)
    const int packTotal = GATES * Hdim;           // 4,194,304
    pack_weight_bf16<<<(packTotal + 255) / 256, 256, 0, stream>>>(Whh0, Whh0p, Hdim);
    pack_weight_bf16<<<(packTotal + 255) / 256, 256, 0, stream>>>(Wih1, Wih1p, Hdim);
    pack_weight_bf16<<<(packTotal + 255) / 256, 256, 0, stream>>>(Whh1, Whh1p, Hdim);
    make_bias<<<(GATES + 255) / 256, 256, 0, stream>>>(bih0, bhh0, bih1, bhh1, b0, b1);
    zero_f32<<<(ZERO_FLOATS + 255u) / 256u, 256, 0, stream>>>((float*)(ws + OFF_H0P0),
                                                              ZERO_FLOATS);

    // sequential recurrence, layers interleaved per timestep
    for (int t = 0; t < Lseq; ++t) {
        int cur = t & 1, nxt = (t + 1) & 1;
        lstm_step<0><<<128, 256, 0, stream>>>(h0p[cur], h0p[nxt], nullptr,
                                              Whh0p, nullptr, b0,
                                              seq, Wih0, c0, nullptr, t);
        lstm_step<1><<<128, 256, 0, stream>>>(h1p[cur], h1p[nxt], h0p[nxt],
                                              Whh1p, Wih1p, b1,
                                              nullptr, nullptr, c1, hf1, t);
    }

    final_linear<<<1, 256, 0, stream>>>(hf1, linW, linb, out);
    (void)in_sizes; (void)n_in; (void)out_size; (void)ws_size;
}
